// TransBlockV2_83485574299848
// MI455X (gfx1250) — compile-verified, matrix operand order checked by reading, not run on hardware
//
#include <hip/hip_runtime.h>
#include <hip/hip_bf16.h>

// ---------------------------------------------------------------------------
// SO(3)-equivariant graph transformer block for MI455X (gfx1250, wave32).
// All GEMMs routed through v_wmma_f32_16x16x32_bf16 on 16-row tiles.
// Weights are stored transposed (N x K) so B-fragments load as 2x b128.
// ---------------------------------------------------------------------------

typedef __attribute__((ext_vector_type(16))) __bf16 v16bf;
typedef __attribute__((ext_vector_type(8)))  __bf16 v8bf;
typedef __attribute__((ext_vector_type(8)))  float  v8f;

#define DEVI __device__ __forceinline__

DEVI float sigmoidf_(float x) {
  return __builtin_amdgcn_rcpf(1.0f + __expf(-x));
}

// --- WMMA tile helpers (16x16x32 bf16 -> f32) ------------------------------
// A: 16(M) x 32(K), row-major with leading dim lda (multiple of 8).
// lane 0-15 : M = lane,    elems 0..7 -> K 0..7,  elems 8..15 -> K 16..23
// lane 16-31: M = lane-16, elems 0..7 -> K 8..15, elems 8..15 -> K 24..31
// Both 8-element groups are K-contiguous -> two aligned 16B loads.
DEVI v16bf load_a32(const __bf16* A, int lda) {
  int ln = threadIdx.x & 31;
  int m  = ln & 15;
  int kb = (ln >> 4) << 3;
  const __bf16* p = A + m * lda + kb;
  v8bf lo = *(const v8bf*)p;
  v8bf hi = *(const v8bf*)(p + 16);
  return __builtin_shufflevector(lo, hi, 0, 1, 2, 3, 4, 5, 6, 7,
                                 8, 9, 10, 11, 12, 13, 14, 15);
}
// B stored transposed: BT is N x K row-major (K-contiguous per column n).
DEVI v16bf load_bt(const __bf16* BT, int K, int n0, int k0) {
  int ln = threadIdx.x & 31;
  int n  = ln & 15;
  int kb = (ln >> 4) << 3;
  const __bf16* p = BT + (size_t)(n0 + n) * K + k0 + kb;
  v8bf lo = *(const v8bf*)p;
  v8bf hi = *(const v8bf*)(p + 16);
  return __builtin_shufflevector(lo, hi, 0, 1, 2, 3, 4, 5, 6, 7,
                                 8, 9, 10, 11, 12, 13, 14, 15);
}

// D-tile = A(16xK) @ B(KxN, tile at col n0), B given transposed. K mult of 32.
DEVI v8f gemm16(const __bf16* A, int lda, const __bf16* BT, int n0, int K) {
  v8f acc;
#pragma unroll
  for (int i = 0; i < 8; ++i) acc[i] = 0.0f;
  for (int k0 = 0; k0 < K; k0 += 32) {
    v16bf a = load_a32(A + k0, lda);
    v16bf b = load_bt(BT, K, n0, k0);
    acc = __builtin_amdgcn_wmma_f32_16x16x32_bf16(false, a, false, b,
                                                  (short)0, acc, false, false);
  }
  return acc;
}
// D tile layout: lane 0-15 -> (M=v, N=lane), lane 16-31 -> (M=v+8, N=lane-16)

// --- utility kernels -------------------------------------------------------
// src: K x Nc row-major (f32) -> dst: Nc x K row-major (bf16), i.e. transposed
__global__ void cvt_t_kernel(const float* __restrict__ s, __bf16* __restrict__ d,
                             int K, int Nc) {
  int idx = blockIdx.x * blockDim.x + threadIdx.x;
  if (idx >= K * Nc) return;
  int k = idx / Nc, n = idx - k * Nc;
  d[(size_t)n * K + k] = (__bf16)s[idx];
}
__global__ void zero_f32_kernel(float* __restrict__ p, int n) {
  int i = blockIdx.x * blockDim.x + threadIdx.x;
  if (i < n) p[i] = 0.0f;
}

// --- enorm: xn = x * rsqrt(groupmean(x^2) + 1e-6) --------------------------
__global__ __launch_bounds__(64) void enorm_kernel(const float* __restrict__ x,
                                                   float* __restrict__ xn) {
  __shared__ float sr[3][64];
  int n = blockIdx.x;
  int c = threadIdx.x;
  const float* xp = x + (size_t)n * 576;
  float v0 = xp[c];
  float p0 = v0 * v0, p1 = 0.f, p2 = 0.f;
  for (int k = 1; k < 4; ++k) { float v = xp[k * 64 + c]; p1 += v * v; }
  for (int k = 4; k < 9; ++k) { float v = xp[k * 64 + c]; p2 += v * v; }
  sr[0][c] = p0; sr[1][c] = p1; sr[2][c] = p2;
  __syncthreads();
  for (int s = 32; s > 0; s >>= 1) {
    if (c < s) {
      sr[0][c] += sr[0][c + s];
      sr[1][c] += sr[1][c + s];
      sr[2][c] += sr[2][c + s];
    }
    __syncthreads();
  }
  float sc0 = rsqrtf(sr[0][0] / 64.f  + 1e-6f);
  float sc1 = rsqrtf(sr[1][0] / 192.f + 1e-6f);
  float sc2 = rsqrtf(sr[2][0] / 320.f + 1e-6f);
  float* op = xn + (size_t)n * 576;
  op[c] = xp[c] * sc0;
  for (int k = 1; k < 4; ++k) op[k * 64 + c] = xp[k * 64 + c] * sc1;
  for (int k = 4; k < 9; ++k) op[k * 64 + c] = xp[k * 64 + c] * sc2;
}

// --- pass 1: attention logits (WMMA) ---------------------------------------
__global__ __launch_bounds__(128) void edge_logits_kernel(
    const float* __restrict__ ed, const float* __restrict__ semb,
    const float* __restrict__ temb, const int* __restrict__ an,
    const int* __restrict__ ei, const __bf16* __restrict__ w1,
    const float* __restrict__ b1, const __bf16* __restrict__ w2,
    const float* __restrict__ b2, const __bf16* __restrict__ wm0,
    const float* __restrict__ lng, const float* __restrict__ lnb,
    const float* __restrict__ adot, const float* __restrict__ xn,
    float* __restrict__ logits, int E) {
  __shared__ int s_src[16], s_tgt[16];
  __shared__ alignas(16) __bf16 s_ef[16 * 128];
  __shared__ alignas(16) __bf16 s_h1[16 * 128];
  __shared__ alignas(16) __bf16 s_rad0[16 * 128];
  __shared__ alignas(16) __bf16 s_xe[16 * 128];
  __shared__ float s_alpha[16 * 256];
  int tid = threadIdx.x, wave = tid >> 5, ln = tid & 31;
  int tn = ln & 15, mb = (ln >> 4) * 8;
  int e0 = blockIdx.x * 16;
  if (tid < 16) { s_src[tid] = ei[e0 + tid]; s_tgt[tid] = ei[E + e0 + tid]; }
  __syncthreads();
  // edge features: [dist(64) | src_emb(32) | tgt_emb(32)]
  for (int idx = tid; idx < 2048; idx += 128) {
    int i = idx >> 7, c = idx & 127;
    float v;
    if (c < 64)      v = ed[(size_t)(e0 + i) * 64 + c];
    else if (c < 96) v = semb[an[s_src[i]] * 32 + (c - 64)];
    else             v = temb[an[s_tgt[i]] * 32 + (c - 96)];
    s_ef[idx] = (__bf16)v;
  }
  __syncthreads();
  for (int t = wave; t < 8; t += 4) {           // h1 = silu(ef@W1+b1)
    int n0 = t * 16;
    v8f acc = gemm16(s_ef, 128, w1, n0, 128);
    float bb = b1[n0 + tn];
#pragma unroll
    for (int v = 0; v < 8; ++v) {
      float z = acc[v] + bb;
      s_h1[(mb + v) * 128 + n0 + tn] = (__bf16)(z * sigmoidf_(z));
    }
  }
  __syncthreads();
  for (int t = wave; t < 8; t += 4) {           // rad[:,0,:] = h1@W2[:,:128]+b2
    int n0 = t * 16;
    v8f acc = gemm16(s_h1, 128, w2, n0, 128);
    float bb = b2[n0 + tn];
#pragma unroll
    for (int v = 0; v < 8; ++v)
      s_rad0[(mb + v) * 128 + n0 + tn] = (__bf16)(acc[v] + bb);
  }
  __syncthreads();
  for (int idx = tid; idx < 2048; idx += 128) { // x_edge0 = concat(xs0,xt0)*rad0
    int i = idx >> 7, c = idx & 127;
    float xv = (c < 64) ? xn[(size_t)s_src[i] * 576 + c]
                        : xn[(size_t)s_tgt[i] * 576 + (c - 64)];
    s_xe[idx] = (__bf16)(xv * (float)s_rad0[idx]);
  }
  __syncthreads();
  for (int t = wave; t < 16; t += 4) {          // alpha = x_edge0 @ Wm0[:,0:256]
    int n0 = t * 16;
    v8f acc = gemm16(s_xe, 128, wm0, n0, 128);
#pragma unroll
    for (int v = 0; v < 8; ++v) s_alpha[(mb + v) * 256 + n0 + tn] = acc[v];
  }
  __syncthreads();
  {                                             // layernorm + slrelu + dot
    int i = tid >> 3, h = tid & 7;
    const float* a = &s_alpha[i * 256 + h * 32];
    float mu = 0.f;
    for (int j = 0; j < 32; ++j) mu += a[j];
    mu *= (1.f / 32.f);
    float var = 0.f;
    for (int j = 0; j < 32; ++j) { float d = a[j] - mu; var += d * d; }
    var *= (1.f / 32.f);
    float rstd = rsqrtf(var + 1e-5f);
    float lg = 0.f;
    for (int j = 0; j < 32; ++j) {
      float z = (a[j] - mu) * rstd * lng[j] + lnb[j];
      float sl = 0.2f * z + 0.8f * z * sigmoidf_(z);
      lg += sl * adot[h * 32 + j];
    }
    logits[(size_t)(e0 + i) * 8 + h] = lg;
  }
}

// --- segment softmax (ordered-uint atomicMax, then exp/atomicAdd) ----------
__global__ void segmax_kernel(const float* __restrict__ logits,
                              const int* __restrict__ ei,
                              unsigned* __restrict__ mord, int E) {
  int idx = blockIdx.x * blockDim.x + threadIdx.x;
  if (idx >= E * 8) return;
  int e = idx >> 3, h = idx & 7;
  int t = ei[E + e];
  int i = __float_as_int(logits[idx]);
  unsigned u = (i < 0) ? ~(unsigned)i : ((unsigned)i | 0x80000000u);
  atomicMax(&mord[t * 8 + h], u);
}
__global__ void segexp_kernel(float* __restrict__ logits,
                              const int* __restrict__ ei,
                              const unsigned* __restrict__ mord,
                              float* __restrict__ ssum, int E) {
  int idx = blockIdx.x * blockDim.x + threadIdx.x;
  if (idx >= E * 8) return;
  int e = idx >> 3, h = idx & 7;
  int t = ei[E + e];
  unsigned u = mord[t * 8 + h];
  int i = (u & 0x80000000u) ? (int)(u & 0x7fffffffu) : ~(int)u;
  float m = __int_as_float(i);
  if (!__builtin_isfinite(m)) m = 0.f;
  float ev = __expf(logits[idx] - m);
  logits[idx] = ev;                 // numerator
  atomicAdd(&ssum[t * 8 + h], ev);
}

// --- pass 2: full edge compute + attn-weighted scatter (WMMA) --------------
__global__ __launch_bounds__(128) void edge_main_kernel(
    const float* __restrict__ ed, const float* __restrict__ semb,
    const float* __restrict__ temb, const float* __restrict__ rl_ij,
    const int* __restrict__ an, const int* __restrict__ ei,
    const __bf16* __restrict__ w1, const float* __restrict__ b1,
    const __bf16* __restrict__ w2, const float* __restrict__ b2,
    const __bf16* __restrict__ wc1, const __bf16* __restrict__ wm0,
    const __bf16* __restrict__ wxj, const __bf16* __restrict__ wc2,
    const float* __restrict__ xn, const float* __restrict__ numer,
    const float* __restrict__ ssum, float* __restrict__ nodeout, int E) {
  __shared__ int s_src[16], s_tgt[16];
  __shared__ float s_rl[16 * 8];
  __shared__ float s_attn[16 * 8];
  __shared__ alignas(16) __bf16 s_ef[16 * 128];
  __shared__ alignas(16) __bf16 s_h1[16 * 128];
  __shared__ alignas(16) __bf16 s_rad[16 * 384];
  __shared__ alignas(16) __bf16 s_xe[16 * 128];
  __shared__ alignas(16) __bf16 s_xsk[16 * 64];
  __shared__ alignas(16) __bf16 s_val[16 * 128];
  __shared__ alignas(16) __bf16 s_gates[16 * 512];
  int tid = threadIdx.x, wave = tid >> 5, ln = tid & 31;
  int tn = ln & 15, mb = (ln >> 4) * 8;
  int e0 = blockIdx.x * 16;
  if (tid < 16) { s_src[tid] = ei[e0 + tid]; s_tgt[tid] = ei[E + e0 + tid]; }
  __syncthreads();
  {                                             // attn + rl to LDS
    int i = tid >> 3, h = tid & 7;
    float nu = numer[(size_t)(e0 + i) * 8 + h];
    s_attn[tid] = nu / (ssum[s_tgt[i] * 8 + h] + 1e-9f);
    s_rl[tid] = rl_ij[(size_t)(e0 + i) * 8 + h];
  }
  for (int idx = tid; idx < 2048; idx += 128) { // edge features
    int i = idx >> 7, c = idx & 127;
    float v;
    if (c < 64)      v = ed[(size_t)(e0 + i) * 64 + c];
    else if (c < 96) v = semb[an[s_src[i]] * 32 + (c - 64)];
    else             v = temb[an[s_tgt[i]] * 32 + (c - 96)];
    s_ef[idx] = (__bf16)v;
  }
  __syncthreads();
  for (int t = wave; t < 8; t += 4) {           // h1
    int n0 = t * 16;
    v8f acc = gemm16(s_ef, 128, w1, n0, 128);
    float bb = b1[n0 + tn];
#pragma unroll
    for (int v = 0; v < 8; ++v) {
      float z = acc[v] + bb;
      s_h1[(mb + v) * 128 + n0 + tn] = (__bf16)(z * sigmoidf_(z));
    }
  }
  __syncthreads();
  for (int t = wave; t < 24; t += 4) {          // rad (E x 3 x 128)
    int n0 = t * 16;
    v8f acc = gemm16(s_h1, 128, w2, n0, 128);
    float bb = b2[n0 + tn];
#pragma unroll
    for (int v = 0; v < 8; ++v)
      s_rad[(mb + v) * 384 + n0 + tn] = (__bf16)(acc[v] + bb);
  }
  __syncthreads();
  for (int idx = tid; idx < 2048; idx += 128) { // x_edge0
    int i = idx >> 7, c = idx & 127;
    float xv = (c < 64) ? xn[(size_t)s_src[i] * 576 + c]
                        : xn[(size_t)s_tgt[i] * 576 + (c - 64)];
    s_xe[idx] = (__bf16)(xv * (float)s_rad[i * 384 + c]);
  }
  __syncthreads();
  for (int t = 16 + wave; t < 24; t += 4) {     // g_scalar -> val0 = silu
    int n0 = t * 16;
    v8f acc = gemm16(s_xe, 128, wm0, n0, 128);
    int c = n0 - 256 + tn;
#pragma unroll
    for (int v = 0; v < 8; ++v) {
      float z = acc[v];
      s_val[(mb + v) * 128 + c] = (__bf16)(z * sigmoidf_(z));
    }
  }
  for (int t = 24 + wave; t < 56; t += 4) {     // gates -> sigmoid
    int n0 = t * 16;
    v8f acc = gemm16(s_xe, 128, wm0, n0, 128);
    int c = n0 - 384 + tn;
#pragma unroll
    for (int v = 0; v < 8; ++v)
      s_gates[(mb + v) * 512 + c] = (__bf16)sigmoidf_(acc[v]);
  }
  __syncthreads();
  for (int t = wave; t < 8; t += 4) {           // vout_0 scatter
    int n0 = t * 16;
    v8f acc = gemm16(s_val, 128, wc2, n0, 128);
    int c = n0 + tn, h = c >> 4;
#pragma unroll
    for (int v = 0; v < 8; ++v) {
      int m = mb + v;
      atomicAdd(&nodeout[((size_t)s_tgt[m] * 9) * 128 + c],
                acc[v] * s_attn[m * 8 + h]);
    }
  }
  for (int k = 1; k < 9; ++k) {
    int l  = (k < 4) ? 1 : 2;
    int l1 = (k < 4) ? 0 : 1;
    __syncthreads();
    for (int idx = tid; idx < 2048; idx += 128) {   // x_edge_k
      int i = idx >> 7, c = idx & 127;
      float xv = (c < 64) ? xn[(size_t)s_src[i] * 576 + k * 64 + c]
                          : xn[(size_t)s_tgt[i] * 576 + k * 64 + (c - 64)];
      s_xe[idx] = (__bf16)(xv * (float)s_rad[i * 384 + l * 128 + c]);
    }
    for (int idx = tid; idx < 1024; idx += 128) {   // xs_k for xj_proj
      int i = idx >> 6, c = idx & 63;
      s_xsk[idx] = (__bf16)xn[(size_t)s_src[i] * 576 + k * 64 + c];
    }
    __syncthreads();
    for (int t = wave; t < 8; t += 4) {             // msg + xjp -> val
      int n0 = t * 16;
      v8f msg = gemm16(s_xe, 128, wc1 + (size_t)l * 16384, n0, 128);
      v8f xjp = gemm16(s_xsk, 64, wxj, n0, 64);
      int c = n0 + tn;
#pragma unroll
      for (int v = 0; v < 8; ++v) {
        int m = mb + v;
        float gd = (float)s_gates[m * 512 + l1 * 128 + c];
        float gt = (float)s_gates[m * 512 + 256 + l1 * 128 + c];
        float vv = msg[v] + s_rl[m * 8 + (k - 1)] * gd + xjp[v] * gt;
        s_val[m * 128 + c] = (__bf16)vv;
      }
    }
    __syncthreads();
    for (int t = wave; t < 8; t += 4) {             // vout_k scatter
      int n0 = t * 16;
      v8f acc = gemm16(s_val, 128, wc2 + (size_t)l * 16384, n0, 128);
      int c = n0 + tn, h = c >> 4;
#pragma unroll
      for (int v = 0; v < 8; ++v) {
        int m = mb + v;
        atomicAdd(&nodeout[((size_t)s_tgt[m] * 9 + k) * 128 + c],
                  acc[v] * s_attn[m * 8 + h]);
      }
    }
  }
}

// --- node update: proj residual + enorm + gated FFN (WMMA) -----------------
__global__ __launch_bounds__(128) void node_kernel(
    const float* __restrict__ x, const float* __restrict__ nodeout,
    const __bf16* __restrict__ wproj, const float* __restrict__ bproj,
    const __bf16* __restrict__ wgate, const float* __restrict__ bgate,
    const __bf16* __restrict__ wffn1, const __bf16* __restrict__ wffn2,
    const float* __restrict__ bffn2, float* __restrict__ out) {
  __shared__ float s_xmid[16 * 576];
  __shared__ alignas(16) __bf16 s_a[16 * 128];
  __shared__ alignas(16) __bf16 s_yn[16 * 64];
  __shared__ float s_g[16 * 128];
  __shared__ alignas(16) __bf16 s_h[16 * 128];
  __shared__ float s_scale[16 * 3];
  int tid = threadIdx.x, wave = tid >> 5, ln = tid & 31;
  int tn = ln & 15, mb = (ln >> 4) * 8;
  int nb = blockIdx.x * 16;
  for (int k = 0; k < 9; ++k) {                 // x_mid = x + so3_linear(node_out)
    int l = (k == 0) ? 0 : ((k < 4) ? 1 : 2);
    __syncthreads();
    for (int idx = tid; idx < 2048; idx += 128) {
      int i = idx >> 7, c = idx & 127;
      s_a[idx] = (__bf16)nodeout[((size_t)(nb + i) * 9 + k) * 128 + c];
    }
    __syncthreads();
    {
      int c0 = wave * 16;                       // 4 col tiles, one per wave
      v8f acc = gemm16(s_a, 128, wproj + (size_t)l * 8192, c0, 128);
      float bb = (k == 0) ? bproj[c0 + tn] : 0.f;
#pragma unroll
      for (int v = 0; v < 8; ++v) {
        int m = mb + v;
        s_xmid[m * 576 + k * 64 + c0 + tn] =
            x[(size_t)(nb + m) * 576 + k * 64 + c0 + tn] + acc[v] + bb;
      }
    }
  }
  __syncthreads();
  if (tid < 48) {                               // enorm scales per (node, l)
    int i = tid / 3, l = tid % 3;
    int k0 = (l == 0) ? 0 : ((l == 1) ? 1 : 4);
    int k1 = (l == 0) ? 1 : ((l == 1) ? 4 : 9);
    float s = 0.f;
    for (int k = k0; k < k1; ++k)
      for (int c = 0; c < 64; ++c) {
        float v = s_xmid[i * 576 + k * 64 + c];
        s += v * v;
      }
    float cnt = (l == 0) ? 64.f : ((l == 1) ? 192.f : 320.f);
    s_scale[i * 3 + l] = rsqrtf(s / cnt + 1e-6f);
  }
  __syncthreads();
  for (int idx = tid; idx < 1024; idx += 128) { // yn[:,0]
    int i = idx >> 6, c = idx & 63;
    s_yn[idx] = (__bf16)(s_xmid[i * 576 + c] * s_scale[i * 3]);
  }
  __syncthreads();
  for (int t = wave; t < 8; t += 4) {           // g = yn0@Wgate + bgate
    int c0 = t * 16;
    v8f acc = gemm16(s_yn, 64, wgate, c0, 64);
    float bb = bgate[c0 + tn];
#pragma unroll
    for (int v = 0; v < 8; ++v) s_g[(mb + v) * 128 + c0 + tn] = acc[v] + bb;
  }
  __syncthreads();
  for (int k = 0; k < 9; ++k) {
    int l = (k == 0) ? 0 : ((k < 4) ? 1 : 2);
    if (k == 0) {                               // h0 = silu(g)  (ffn1@k0 is dead)
      for (int idx = tid; idx < 2048; idx += 128) {
        float z = s_g[idx];
        s_h[idx] = (__bf16)(z * sigmoidf_(z));
      }
    } else {
      for (int idx = tid; idx < 1024; idx += 128) {
        int i = idx >> 6, c = idx & 63;
        s_yn[idx] = (__bf16)(s_xmid[i * 576 + k * 64 + c] * s_scale[i * 3 + l]);
      }
      __syncthreads();
      for (int t = wave; t < 8; t += 4) {       // h_k = (yn_k@Wffn1)*sigmoid(g)
        int c0 = t * 16;
        v8f acc = gemm16(s_yn, 64, wffn1 + (size_t)l * 8192, c0, 64);
#pragma unroll
        for (int v = 0; v < 8; ++v) {
          int m = mb + v;
          s_h[m * 128 + c0 + tn] =
              (__bf16)(acc[v] * sigmoidf_(s_g[m * 128 + c0 + tn]));
        }
      }
    }
    __syncthreads();
    {                                           // out = x_mid + h@Wffn2 (+b@k0)
      int c0 = wave * 16;
      v8f acc = gemm16(s_h, 128, wffn2 + (size_t)l * 8192, c0, 128);
      float bb = (k == 0) ? bffn2[c0 + tn] : 0.f;
#pragma unroll
      for (int v = 0; v < 8; ++v) {
        int m = mb + v;
        out[(size_t)(nb + m) * 576 + k * 64 + c0 + tn] =
            s_xmid[m * 576 + k * 64 + c0 + tn] + acc[v] + bb;
      }
    }
    __syncthreads();
  }
}

// ---------------------------------------------------------------------------
extern "C" void kernel_launch(void* const* d_in, const int* in_sizes, int n_in,
                              void* d_out, int out_size, void* d_ws, size_t ws_size,
                              hipStream_t stream) {
  (void)in_sizes; (void)n_in; (void)out_size; (void)ws_size;
  constexpr int N = 10000, E = 80000;
  const float* x    = (const float*)d_in[0];
  const float* ed   = (const float*)d_in[1];
  const float* rl   = (const float*)d_in[2];
  const float* semb = (const float*)d_in[3];
  const float* temb = (const float*)d_in[4];
  const float* W1   = (const float*)d_in[5];
  const float* b1   = (const float*)d_in[6];
  const float* W2   = (const float*)d_in[7];
  const float* b2   = (const float*)d_in[8];
  const float* Wc1  = (const float*)d_in[9];
  const float* Wm0  = (const float*)d_in[10];
  const float* lng  = (const float*)d_in[11];
  const float* lnb  = (const float*)d_in[12];
  const float* adot = (const float*)d_in[13];
  const float* Wxj  = (const float*)d_in[14];
  const float* Wc2  = (const float*)d_in[15];
  const float* Wpr  = (const float*)d_in[16];
  const float* bpr  = (const float*)d_in[17];
  const float* Wg   = (const float*)d_in[18];
  const float* bg   = (const float*)d_in[19];
  const float* Wf1  = (const float*)d_in[20];
  // d_in[21] = b_ffn1 is dead in the reference (h[:,0] overwritten by silu(g))
  const float* Wf2  = (const float*)d_in[22];
  const float* bf2  = (const float*)d_in[23];
  const int* an     = (const int*)d_in[24];
  const int* ei     = (const int*)d_in[25];
  float* out = (float*)d_out;

  char* ws = (char*)d_ws;
  size_t off = 0;
  auto alloc = [&](size_t bytes) -> void* {
    off = (off + 255) & ~(size_t)255;
    void* p = ws + off;
    off += bytes;
    return p;
  };
  __bf16* w1t  = (__bf16*)alloc(16384  * 2);
  __bf16* w2t  = (__bf16*)alloc(49152  * 2);
  __bf16* wc1t = (__bf16*)alloc(49152  * 2);
  __bf16* wm0t = (__bf16*)alloc(114688 * 2);
  __bf16* wxjt = (__bf16*)alloc(8192   * 2);
  __bf16* wc2t = (__bf16*)alloc(49152  * 2);
  __bf16* wprt = (__bf16*)alloc(24576  * 2);
  __bf16* wgt  = (__bf16*)alloc(8192   * 2);
  __bf16* wf1t = (__bf16*)alloc(24576  * 2);
  __bf16* wf2t = (__bf16*)alloc(24576  * 2);
  float*    xn     = (float*)alloc((size_t)N * 576 * 4);
  float*    logits = (float*)alloc((size_t)E * 8 * 4);
  unsigned* mord   = (unsigned*)alloc((size_t)N * 8 * 4);
  float*    ssum   = (float*)alloc((size_t)N * 8 * 4);
  float*    nout   = (float*)alloc((size_t)N * 1152 * 4);

  // transpose-convert weights: (K x N) f32 -> (N x K) bf16
  auto cvtT = [&](const float* s, __bf16* d, int K, int Nc) {
    int n = K * Nc;
    cvt_t_kernel<<<(n + 255) / 256, 256, 0, stream>>>(s, d, K, Nc);
  };
  cvtT(W1, w1t, 128, 128);
  cvtT(W2, w2t, 128, 384);
  cvtT(Wm0, wm0t, 128, 896);
  cvtT(Wxj, wxjt, 64, 128);
  cvtT(Wg, wgt, 64, 128);
  for (int l = 0; l < 3; ++l) {
    cvtT(Wc1 + l * 16384, wc1t + l * 16384, 128, 128);
    cvtT(Wc2 + l * 16384, wc2t + l * 16384, 128, 128);
    cvtT(Wpr + l * 8192,  wprt + l * 8192,  128, 64);
    cvtT(Wf1 + l * 8192,  wf1t + l * 8192,  64, 128);
    cvtT(Wf2 + l * 8192,  wf2t + l * 8192,  128, 64);
  }

  zero_f32_kernel<<<(N * 8 + 255) / 256, 256, 0, stream>>>((float*)mord, N * 8);
  zero_f32_kernel<<<(N * 8 + 255) / 256, 256, 0, stream>>>(ssum, N * 8);
  zero_f32_kernel<<<(N * 1152 + 255) / 256, 256, 0, stream>>>(nout, N * 1152);

  enorm_kernel<<<N, 64, 0, stream>>>(x, xn);
  edge_logits_kernel<<<E / 16, 128, 0, stream>>>(ed, semb, temb, an, ei, w1t, b1,
                                                 w2t, b2, wm0t, lng, lnb, adot,
                                                 xn, logits, E);
  segmax_kernel<<<(E * 8 + 255) / 256, 256, 0, stream>>>(logits, ei, mord, E);
  segexp_kernel<<<(E * 8 + 255) / 256, 256, 0, stream>>>(logits, ei, mord, ssum, E);
  edge_main_kernel<<<E / 16, 128, 0, stream>>>(ed, semb, temb, rl, an, ei, w1t, b1,
                                               w2t, b2, wc1t, wm0t, wxjt, wc2t,
                                               xn, logits, ssum, nout, E);
  node_kernel<<<N / 16, 128, 0, stream>>>(x, nout, wprt, bpr, wgt, bg, wf1t,
                                          wf2t, bf2, out);
}
